// SimpleTransformer_22660247454432
// MI455X (gfx1250) — matrix operand order repaired and checked by reading, not compiled, not run on hardware
//
#include <hip/hip_runtime.h>
#include <hip/hip_bf16.h>
#include <cstdint>
#include <cstddef>

// ---------------- model constants ----------------
#define BSZ   256
#define NPLY  8
#define DPL   160
#define DFF   1024
#define NBLK  12
#define NWEAP 139
#define NRANK 20
#define NLOB  8
#define NMOD  5
#define NSTG  23

typedef float  v8f   __attribute__((ext_vector_type(8)));
typedef __bf16 v16bf __attribute__((ext_vector_type(16)));
typedef unsigned int u32x4 __attribute__((ext_vector_type(4)));
typedef int          i32x4 __attribute__((ext_vector_type(4)));
typedef int          i32x8 __attribute__((ext_vector_type(8)));

#if defined(__gfx1250__) && __has_builtin(__builtin_amdgcn_tensor_load_to_lds)
#define HAVE_TDM 1
#else
#define HAVE_TDM 0
#endif

// ---------------- TDM helpers ----------------
// Issue a 2D tile load (tileW cols x 32 rows, f32) from a row-major matrix with
// row stride `rowStride` (elements) into LDS at byte offset ldsOff.
// D# packing per CDNA5 ISA (group0: count/lds/global/type,
// group1: data_size=4B, tensor_dim0/1, tile_dim0/1, dim0 stride).
static __device__ __forceinline__ void tdm_issue(const float* gsrc, unsigned ldsOff,
                                                 int tileW, int rowStride) {
#if HAVE_TDM
  unsigned long long ga = (unsigned long long)(uintptr_t)gsrc;
  u32x4 g0;
  g0[0] = 1u;                                   // count=1, user descriptor
  g0[1] = ldsOff;                               // lds_addr (bytes)
  g0[2] = (unsigned)(ga & 0xFFFFFFFFull);       // global_addr[31:0]
  g0[3] = (unsigned)((ga >> 32) & 0x01FFFFFFu)  // global_addr[56:32]
        | (2u << 30);                           // type = 2 ("image")
  unsigned td0 = (unsigned)rowStride;           // tensor_dim0
  unsigned td1 = (1u << 20);                    // tensor_dim1 (large; in-bounds use)
  i32x8 g1;
  g1[0] = (int)(2u << 16);                      // data_size = 4 bytes
  g1[1] = (int)((td0 & 0xFFFFu) << 16);         // tensor_dim0 lo16 at bits[63:48]
  g1[2] = (int)((td0 >> 16) | ((td1 & 0xFFFFu) << 16));
  g1[3] = (int)((td1 >> 16) | ((unsigned)tileW << 16));   // tile_dim0
  g1[4] = (int)32u;                             // tile_dim1 = 32, tile_dim2 = 0
  g1[5] = (int)td0;                             // tensor_dim0_stride lo32 (elements)
  g1[6] = 0;                                    // stride hi bits
  g1[7] = 0;
  i32x4 z4; z4[0] = z4[1] = z4[2] = z4[3] = 0;
#if __clang_major__ >= 23
  i32x8 z8; z8[0]=z8[1]=z8[2]=z8[3]=z8[4]=z8[5]=z8[6]=z8[7]=0;
  __builtin_amdgcn_tensor_load_to_lds(g0, g1, z4, z4, z8, 0);
#else
  __builtin_amdgcn_tensor_load_to_lds(g0, g1, z4, z4, 0);
#endif
#else
  (void)gsrc; (void)ldsOff; (void)tileW; (void)rowStride;
#endif
}

static __device__ __forceinline__ void twait0() {
#if __has_builtin(__builtin_amdgcn_s_wait_tensorcnt)
  __builtin_amdgcn_s_wait_tensorcnt(0);
#endif
}
static __device__ __forceinline__ void twait1() {
#if __has_builtin(__builtin_amdgcn_s_wait_tensorcnt)
  __builtin_amdgcn_s_wait_tensorcnt(1);
#endif
}

static __device__ __forceinline__ v16bf pack16(float4 a0, float4 a1, float4 a2, float4 a3) {
  v16bf r;
  r[0]=(__bf16)a0.x;  r[1]=(__bf16)a0.y;  r[2]=(__bf16)a0.z;  r[3]=(__bf16)a0.w;
  r[4]=(__bf16)a1.x;  r[5]=(__bf16)a1.y;  r[6]=(__bf16)a1.z;  r[7]=(__bf16)a1.w;
  r[8]=(__bf16)a2.x;  r[9]=(__bf16)a2.y;  r[10]=(__bf16)a2.z; r[11]=(__bf16)a2.w;
  r[12]=(__bf16)a3.x; r[13]=(__bf16)a3.y; r[14]=(__bf16)a3.z; r[15]=(__bf16)a3.w;
  return r;
}

// ---------------- WMMA GEMM: Y[p] = op(X[p] @ W[p] + bias[p]) ----------------
// X: [P][BSZ][KT] f32, W: [KT][NTOT] f32 per player, Y: [P][BSZ][NTOT] f32.
// WG = 256 threads = 8 waves; WG tile = 128 rows x NT cols; wave = 16-row strip.
// flags: 1 = relu(after bias), 2 = add residual R (after relu) -> matches
// reference order (FFN1: relu only; O-proj/FFN2: residual only).
template<int KT, int NT>
__global__ __launch_bounds__(256) void cell_gemm(
    const float* __restrict__ X, const float* __restrict__ W,
    const float* __restrict__ Bias, const float* __restrict__ R,
    float* __restrict__ Y, int NTOT, long long wStride, long long bStride, int flags)
{
  constexpr int KSTEPS = KT / 32;
  constexpr int NF     = NT / 16;
  constexpr int TILE   = 32 * NT;

  __shared__ __align__(16) unsigned char smem[2 * TILE * 4 + 2 * TILE * 2];
  float*  ldsF = (float*)smem;                    // [2][TILE] f32 staging (TDM dest)
  __bf16* ldsB = (__bf16*)(smem + 2 * TILE * 4);  // [2][TILE] bf16 fragments

  const int p  = blockIdx.z;
  const int m0 = blockIdx.x * 128;
  const int n0 = blockIdx.y * NT;
  const float* Xp = X + (size_t)p * BSZ * KT;
  const float* Wp = W + (size_t)p * wStride;
  const float* Bp = Bias + (size_t)p * bStride;
  const float* Rp = R + (size_t)p * BSZ * NTOT;
  float*       Yp = Y + (size_t)p * BSZ * NTOT;

  const int tid   = (int)threadIdx.x;
  const int lane  = tid & 31;
  const int wave  = tid >> 5;
  const int rowA  = m0 + wave * 16 + (lane & 15);
  const int halfA = lane >> 4;

  v8f acc[NF];
  #pragma unroll
  for (int i = 0; i < NF; ++i) {
    #pragma unroll
    for (int j = 0; j < 8; ++j) acc[i][j] = 0.0f;
  }

#if HAVE_TDM
  if (tid < 32) tdm_issue(Wp + n0, 0u, NT, NTOT);    // prefetch tile 0 (buf 0)
#endif

  for (int ks = 0; ks < KSTEPS; ++ks) {
    const int buf = ks & 1;
#if HAVE_TDM
    if (tid < 32) {
      if (ks + 1 < KSTEPS) {   // issue next tile into other buffer, then wait cur
        tdm_issue(Wp + (size_t)(ks + 1) * 32 * NTOT + n0,
                  (unsigned)(((ks + 1) & 1) * TILE * 4), NT, NTOT);
        twait1();
      } else {
        twait0();
      }
    }
#else
    for (int i = tid; i < TILE; i += 256) {
      int kk = i / NT, nn = i % NT;
      ldsF[buf * TILE + i] = Wp[(size_t)(ks * 32 + kk) * NTOT + n0 + nn];
    }
#endif
    __syncthreads();

    // f32 tile -> bf16 tile (cooperative)
    for (int i = tid; i < TILE; i += 256)
      ldsB[buf * TILE + i] = (__bf16)ldsF[buf * TILE + i];
    __syncthreads();

    // A fragment (16x32 bf16): lane = M row, K pairs per VGPR, K+-8 by lane half
    const float* xr = Xp + (size_t)rowA * KT + ks * 32 + halfA * 8;
    float4 a0 = *(const float4*)(xr);
    float4 a1 = *(const float4*)(xr + 4);
    float4 a2 = *(const float4*)(xr + 16);
    float4 a3 = *(const float4*)(xr + 20);
    v16bf af = pack16(a0, a1, a2, a3);

    // B fragments (32x16 bf16): lane = N col, K in same order as A (K+-8 by lane half)
    const __bf16* brow = ldsB + buf * TILE + (lane & 15);
    #pragma unroll
    for (int nf = 0; nf < NF; ++nf) {
      v16bf bf;
      #pragma unroll
      for (int s = 0; s < 16; ++s) {
        const int kk = (s < 8) ? (halfA * 8 + s) : (16 + halfA * 8 + (s - 8));
        bf[s] = brow[kk * NT + nf * 16];
      }
      acc[nf] = __builtin_amdgcn_wmma_f32_16x16x32_bf16(
          false, af, false, bf, (short)0, acc[nf], false, false);
    }
    __syncthreads();
  }

  // Epilogue: bias (+relu) (+residual), store f32. C layout: lane = N col,
  // VGPR r = M row (r or 8+r by lane half).
  #pragma unroll
  for (int nf = 0; nf < NF; ++nf) {
    const int col = n0 + nf * 16 + (lane & 15);
    const float bv = Bp[col];
    #pragma unroll
    for (int r = 0; r < 8; ++r) {
      const int row = m0 + wave * 16 + halfA * 8 + r;
      float v = acc[nf][r] + bv;
      if (flags & 1) v = fmaxf(v, 0.0f);
      if (flags & 2) v += Rp[(size_t)row * NTOT + col];
      Yp[(size_t)row * NTOT + col] = v;
    }
  }
}

// ---------------- per-sample outer-product softmax attention ----------------
// o_i = sum_j softmax_j(q_i k_j / sqrt(d)) v_j   (d = 160, per sample)
__global__ __launch_bounds__(256) void attn_kernel(
    const float* __restrict__ Q, const float* __restrict__ K,
    const float* __restrict__ V, float* __restrict__ O)
{
  __shared__ float sk[8][DPL];
  __shared__ float sv[8][DPL];
  const int lane = threadIdx.x & 31;
  const int wave = threadIdx.x >> 5;
  const int b = blockIdx.x * 8 + wave;
  const int p = blockIdx.y;
  const size_t base = ((size_t)p * BSZ + b) * DPL;

  for (int i = lane; i < DPL; i += 32) {
    sk[wave][i] = K[base + i];
    sv[wave][i] = V[base + i];
  }
  __syncthreads();

  const float scale = rsqrtf((float)DPL);
  for (int i = lane; i < DPL; i += 32) {
    const float q = Q[base + i] * scale;
    float m = -3.402823466e38f;
    for (int j = 0; j < DPL; ++j) m = fmaxf(m, q * sk[wave][j]);
    float den = 0.0f, num = 0.0f;
    for (int j = 0; j < DPL; ++j) {
      const float e = __expf(q * sk[wave][j] - m);
      den += e;
      num += e * sv[wave][j];
    }
    O[base + i] = num / den;
  }
}

// ---------------- small (lobby/mode/stage) cells, one workgroup per sample ----
struct CellP {
  const float *Wq, *bq, *Wk, *bk, *Wv, *bv, *Wo, *bo, *W1, *b1, *W2, *b2;
};

static __device__ void small_cell(float* xg, int d, CellP c,
                                  float* x, float* q, float* k, float* v,
                                  float* oatt, float* x2, float* part)
{
  const int t = (int)threadIdx.x;
  const int b = (int)blockIdx.x;
  if (t < d) x[t] = xg[b * d + t];
  __syncthreads();

  if (t < d) {
    float qq = c.bq[t], kk = c.bk[t], vv = c.bv[t];
    for (int i = 0; i < d; ++i) {
      const float xi = x[i];
      qq += xi * c.Wq[i * d + t];
      kk += xi * c.Wk[i * d + t];
      vv += xi * c.Wv[i * d + t];
    }
    q[t] = qq; k[t] = kk; v[t] = vv;
  }
  __syncthreads();

  if (t < d) {
    const float sc = rsqrtf((float)d);
    const float qi = q[t] * sc;
    float m = -3.402823466e38f;
    for (int j = 0; j < d; ++j) m = fmaxf(m, qi * k[j]);
    float den = 0.0f, num = 0.0f;
    for (int j = 0; j < d; ++j) {
      const float e = __expf(qi * k[j] - m);
      den += e; num += e * v[j];
    }
    oatt[t] = num / den;
  }
  __syncthreads();

  if (t < d) {
    float o = c.bo[t];
    for (int j = 0; j < d; ++j) o += oatt[j] * c.Wo[j * d + t];
    x2[t] = x[t] + o;
  }
  __syncthreads();

  // FFN: each thread owns dff/128 hidden units, accumulates partial y[d]
  float yloc[NSTG];
  for (int i = 0; i < d; ++i) yloc[i] = 0.0f;
  for (int j = t; j < DFF; j += 128) {
    float h = c.b1[j];
    for (int i = 0; i < d; ++i) h += x2[i] * c.W1[i * DFF + j];
    h = fmaxf(h, 0.0f);
    for (int i = 0; i < d; ++i) yloc[i] += h * c.W2[j * d + i];
  }
  for (int i = 0; i < d; ++i) part[t * (NSTG + 1) + i] = yloc[i];
  __syncthreads();

  if (t < d) {
    float s = x2[t] + c.b2[t];
    for (int tt = 0; tt < 128; ++tt) s += part[tt * (NSTG + 1) + t];
    xg[b * d + t] = s;
  }
  __syncthreads();
}

__global__ __launch_bounds__(128) void small_cells_kernel(
    CellP cl, CellP cm, CellP cs, float* xl, float* xm, float* xs)
{
  __shared__ float x[32], q[32], k[32], v[32], oatt[32], x2[32];
  __shared__ float part[128 * (NSTG + 1)];
  small_cell(xl, NLOB, cl, x, q, k, v, oatt, x2, part);
  small_cell(xm, NMOD, cm, x, q, k, v, oatt, x2, part);
  small_cell(xs, NSTG, cs, x, q, k, v, oatt, x2, part);
}

// ---------------- embeddings + layernorm ----------------
static __device__ void ln_store(const float* row, int n, const float* g,
                                const float* bb, float* dst)
{
  float mu = 0.0f;
  for (int i = 0; i < n; ++i) mu += row[i];
  mu /= (float)n;
  float var = 0.0f;
  for (int i = 0; i < n; ++i) { const float d = row[i] - mu; var += d * d; }
  var /= (float)n;
  const float inv = rsqrtf(var + 1e-5f);
  for (int i = 0; i < n; ++i) dst[i] = (row[i] - mu) * inv * g[i] + bb[i];
}

__global__ __launch_bounds__(32) void embed_kernel(
    const int* __restrict__ weapons, const int* __restrict__ ranks,
    const float* __restrict__ levels, const int* __restrict__ lobby_idx,
    const int* __restrict__ mode_idx, const int* __restrict__ stage_idx,
    const float* wt, const float* wg, const float* wb,
    const float* rt, const float* rg, const float* rb,
    const float* lt, const float* lg, const float* lb,
    const float* mt, const float* mg, const float* mb,
    const float* st, const float* sg, const float* sb,
    float* X, float* lobbyS, float* modeS, float* stageS)
{
  const int b = (int)blockIdx.x;
  const int t = (int)threadIdx.x;
  if (t < NPLY) {
    float* xp = X + ((size_t)t * BSZ + b) * DPL;
    const int wi = weapons[b * NPLY + t];
    const int ri = ranks[b * NPLY + t];
    ln_store(wt + (size_t)wi * NWEAP, NWEAP, wg, wb, xp);
    ln_store(rt + (size_t)ri * NRANK, NRANK, rg, rb, xp + NWEAP);
    xp[NWEAP + NRANK] = levels[b * NPLY + t];
  } else if (t == 8) {
    ln_store(lt + (size_t)lobby_idx[b] * NLOB, NLOB, lg, lb, lobbyS + b * NLOB);
  } else if (t == 9) {
    ln_store(mt + (size_t)mode_idx[b] * NMOD, NMOD, mg, mb, modeS + b * NMOD);
  } else if (t == 10) {
    ln_store(st + (size_t)stage_idx[b] * NSTG, NSTG, sg, sb, stageS + b * NSTG);
  }
}

// ---------------- output head ----------------
__global__ __launch_bounds__(128) void head_kernel(
    const float* __restrict__ lob, const float* __restrict__ mod,
    const float* __restrict__ stg, const float* __restrict__ X,
    const float* __restrict__ Wout, const float* __restrict__ bout,
    float* __restrict__ out)
{
  const int b = (int)(blockIdx.x * blockDim.x + threadIdx.x);
  if (b >= BSZ) return;
  float o0 = bout[0], o1 = bout[1];
  int f = 0;
  auto acc = [&](float v) { o0 += v * Wout[f * 2 + 0]; o1 += v * Wout[f * 2 + 1]; ++f; };
  for (int i = 0; i < NLOB; ++i) acc(lob[b * NLOB + i]);
  for (int i = 0; i < NMOD; ++i) acc(mod[b * NMOD + i]);
  for (int i = 0; i < NSTG; ++i) acc(stg[b * NSTG + i]);
  for (int p = 0; p < NPLY; ++p)
    for (int i = 0; i < DPL; ++i) acc(X[((size_t)p * BSZ + b) * DPL + i]);
  out[b * 2 + 0] = o0;
  out[b * 2 + 1] = o1;
}

// ---------------- host launcher ----------------
extern "C" void kernel_launch(void* const* d_in, const int* in_sizes, int n_in,
                              void* d_out, int out_size, void* d_ws, size_t ws_size,
                              hipStream_t stream)
{
  (void)in_sizes; (void)n_in; (void)out_size; (void)ws_size;
  auto F = [&](int i) { return (const float*)d_in[i]; };
  auto I = [&](int i) { return (const int*)d_in[i]; };

  // setup_inputs() insertion order flattening:
  // 0 weapons, 1 ranks, 2 levels, 3 lobby_mode, 4 mode, 5 stage,
  // 6-8 weapon{t,g,b}, 9-11 rank, 12-14 lobby, 15-17 mode, 18-20 stage,
  // 21-32 blocks.lobby{Wq,bq,Wk,bk,Wv,bv,Wo,bo,W1,b1,W2,b2},
  // 33-44 blocks.mode, 45-56 blocks.stage, 57-68 blocks.player, 69 Wout, 70 bout.

  // workspace carve-up (f32)
  float* w = (float*)d_ws;
  const size_t PSZ = (size_t)NPLY * BSZ * DPL;   // 327,680
  float* Xst   = w;  w += PSZ;
  float* Qb    = w;  w += PSZ;
  float* Kb    = w;  w += PSZ;
  float* Vb    = w;  w += PSZ;
  float* Oatt  = w;  w += PSZ;
  float* X2    = w;  w += PSZ;
  float* H     = w;  w += (size_t)NPLY * BSZ * DFF;
  float* lobbyS = w; w += (size_t)BSZ * NLOB;
  float* modeS  = w; w += (size_t)BSZ * NMOD;
  float* stageS = w; w += (size_t)BSZ * NSTG;

  embed_kernel<<<BSZ, 32, 0, stream>>>(
      I(0), I(1), F(2), I(3), I(4), I(5),
      F(6), F(7), F(8), F(9), F(10), F(11), F(12), F(13), F(14),
      F(15), F(16), F(17), F(18), F(19), F(20),
      Xst, lobbyS, modeS, stageS);

  const dim3 g160(2, 1, NPLY);
  const dim3 gff1(2, DFF / 128, NPLY);

  for (int blk = 0; blk < NBLK; ++blk) {
    const size_t wOff  = (size_t)blk * NPLY * DPL * DPL;
    const size_t bOff  = (size_t)blk * NPLY * DPL;
    const size_t w1Off = (size_t)blk * NPLY * DPL * DFF;
    const size_t b1Off = (size_t)blk * NPLY * DFF;
    const size_t w2Off = (size_t)blk * NPLY * DFF * DPL;

    const float* pWq = F(57) + wOff;  const float* pbq = F(58) + bOff;
    const float* pWk = F(59) + wOff;  const float* pbk = F(60) + bOff;
    const float* pWv = F(61) + wOff;  const float* pbv = F(62) + bOff;
    const float* pWo = F(63) + wOff;  const float* pbo = F(64) + bOff;
    const float* pW1 = F(65) + w1Off; const float* pb1 = F(66) + b1Off;
    const float* pW2 = F(67) + w2Off; const float* pb2 = F(68) + bOff;

    cell_gemm<DPL, DPL><<<g160, 256, 0, stream>>>(
        Xst, pWq, pbq, Xst, Qb, DPL, (long long)DPL * DPL, DPL, 0);
    cell_gemm<DPL, DPL><<<g160, 256, 0, stream>>>(
        Xst, pWk, pbk, Xst, Kb, DPL, (long long)DPL * DPL, DPL, 0);
    cell_gemm<DPL, DPL><<<g160, 256, 0, stream>>>(
        Xst, pWv, pbv, Xst, Vb, DPL, (long long)DPL * DPL, DPL, 0);

    attn_kernel<<<dim3(BSZ / 8, NPLY), 256, 0, stream>>>(Qb, Kb, Vb, Oatt);

    cell_gemm<DPL, DPL><<<g160, 256, 0, stream>>>(
        Oatt, pWo, pbo, Xst, X2, DPL, (long long)DPL * DPL, DPL, 2);
    cell_gemm<DPL, 128><<<gff1, 256, 0, stream>>>(
        X2, pW1, pb1, X2, H, DFF, (long long)DPL * DFF, DFF, 1);
    cell_gemm<DFF, DPL><<<g160, 256, 0, stream>>>(
        H, pW2, pb2, X2, Xst, DPL, (long long)DFF * DPL, DPL, 2);

    auto mkcell = [&](int base, int d) {
      CellP c;
      c.Wq = F(base + 0) + (size_t)blk * d * d;  c.bq = F(base + 1) + (size_t)blk * d;
      c.Wk = F(base + 2) + (size_t)blk * d * d;  c.bk = F(base + 3) + (size_t)blk * d;
      c.Wv = F(base + 4) + (size_t)blk * d * d;  c.bv = F(base + 5) + (size_t)blk * d;
      c.Wo = F(base + 6) + (size_t)blk * d * d;  c.bo = F(base + 7) + (size_t)blk * d;
      c.W1 = F(base + 8) + (size_t)blk * d * DFF;
      c.b1 = F(base + 9) + (size_t)blk * DFF;
      c.W2 = F(base + 10) + (size_t)blk * DFF * d;
      c.b2 = F(base + 11) + (size_t)blk * d;
      return c;
    };
    small_cells_kernel<<<BSZ, 128, 0, stream>>>(
        mkcell(21, NLOB), mkcell(33, NMOD), mkcell(45, NSTG),
        lobbyS, modeS, stageS);
  }

  head_kernel<<<(BSZ + 127) / 128, 128, 0, stream>>>(
      lobbyS, modeS, stageS, Xst, F(69), F(70), (float*)d_out);
}
